// DEC_51342039056886
// MI455X (gfx1250) — matrix operand order repaired and checked by reading, not compiled, not run on hardware
//
#include <hip/hip_runtime.h>
#include <hip/hip_bf16.h>

typedef __attribute__((ext_vector_type(16))) _Float16 v16h;
typedef __attribute__((ext_vector_type(8)))  _Float16 v8h;
typedef __attribute__((ext_vector_type(8)))  float    v8f;

#define B_SZ 128
#define L_SZ 512
#define H_SZ 256
#define H3   768
#define NWIN 5

static __device__ __forceinline__ float sigmoidf_(float x){ return 1.0f/(1.0f+__expf(-x)); }

static __device__ __forceinline__ v8f zero8(){
  v8f z;
#pragma unroll
  for (int i=0;i<8;i++) z[i]=0.0f;
  return z;
}

// ---- WMMA fragment loaders (CDNA5 v_wmma_f32_16x16x32_f16 layouts) ----
// A (16x32 f16): M = lane%16; lanes 0-15: V0-3 -> K 0..7, V4-7 -> K 16..23;
// lanes 16-31 add +8 to K.  Per lane: two contiguous 8-half chunks 16 halfs apart.
static __device__ __forceinline__ v16h load_a_frag(const _Float16* lds, int stride, int ktile){
  int lane = threadIdx.x & 31;
  int m   = lane & 15;
  int sel = lane >> 4;
  const _Float16* p = lds + m*stride + ktile*32 + sel*8;
  v8h lo = *(const v8h*)p;
  v8h hi = *(const v8h*)(p + 16);
  v16h r;
#pragma unroll
  for (int i=0;i<8;i++){ r[i]=lo[i]; r[8+i]=hi[i]; }
  return r;
}

// B (32x16 f16): pre-tiled in workspace, one fragment = 32 lanes x 16 contiguous halfs.
static __device__ __forceinline__ v16h load_b_frag(const _Float16* __restrict__ g, int frag){
  int lane = threadIdx.x & 31;
  return *(const v16h*)(g + (size_t)frag*512 + lane*16);
}

// ---- weight conversion: f32 row-major w[N][K] -> f16 fragment-tiled B for D = A x w^T ----
__global__ void __launch_bounds__(256) convert_tiled_kernel(
    const float* __restrict__ w, _Float16* __restrict__ dst, int N, int K){
  int o = blockIdx.x*blockDim.x + threadIdx.x;
  if (o >= N*K) return;
  int ktiles = K >> 5;
  int frag = o >> 9;
  int lane = (o >> 4) & 31;
  int e    = o & 15;
  int ntile = frag / ktiles;
  int ktile = frag - ntile*ktiles;
  int n = ntile*16 + (lane & 15);
  int k = ktile*32 + ((lane >> 4) << 3) + (e & 7) + ((e >> 3) << 4);
  dst[o] = (_Float16)w[n*K + k];
}

// ---- GRU layer 0: input dim 3 (VALU x-projection), hidden GEMM via WMMA ----
__global__ void __launch_bounds__(256) gru_l0_kernel(
    const float* __restrict__ x,
    const float* __restrict__ wih0_a, const float* __restrict__ wih0_b,
    const _Float16* __restrict__ whh_a, const _Float16* __restrict__ whh_b,
    const float* __restrict__ bih_a, const float* __restrict__ bhh_a,
    const float* __restrict__ bih_b, const float* __restrict__ bhh_b,
    _Float16* __restrict__ out_a, _Float16* __restrict__ out_b2)
{
  const int g = blockIdx.y;
  const float* wih0   = g ? wih0_b : wih0_a;
  const _Float16* whh = g ? whh_b  : whh_a;
  const float* bih    = g ? bih_b  : bih_a;
  const float* bhh    = g ? bhh_b  : bhh_a;
  _Float16* out       = g ? out_b2 : out_a;

  __shared__ _Float16 hp[16*H3];       // h @ whh^T, f16
  __shared__ _Float16 hcur[16*272];    // hidden state, padded stride
  __shared__ float sbih[H3];
  __shared__ float sbhh[H3];
  __shared__ float swih[H3*3];
  __shared__ float sx[16*4];

  const int tid  = threadIdx.x;
  const int wave = tid >> 5;
  const int lane = tid & 31;
  const int b0   = blockIdx.x * 16;

  for (int i = tid; i < H3; i += 256){ sbih[i] = bih[i]; sbhh[i] = bhh[i]; }
  for (int i = tid; i < H3*3; i += 256) swih[i] = wih0[i];
  for (int i = tid; i < 16*272; i += 256) hcur[i] = (_Float16)0.0f;
  __syncthreads();

  const int j = tid;
  const float w_r0 = swih[j*3+0],        w_r1 = swih[j*3+1],        w_r2 = swih[j*3+2];
  const float w_z0 = swih[(256+j)*3+0],  w_z1 = swih[(256+j)*3+1],  w_z2 = swih[(256+j)*3+2];
  const float w_n0 = swih[(512+j)*3+0],  w_n1 = swih[(512+j)*3+1],  w_n2 = swih[(512+j)*3+2];
  const float bi_r = sbih[j],     bi_z = sbih[256+j],  bi_n = sbih[512+j];
  const float bh_r = sbhh[j],     bh_z = sbhh[256+j],  bh_n = sbhh[512+j];

  for (int t = 0; t < L_SZ; ++t){
    if (tid < 48){
      int m = tid/3, c = tid - m*3;
      const float* xp = x + ((size_t)(b0+m)*L_SZ + t)*3 + c;
      sx[m*4+c] = *xp;
      __builtin_prefetch(xp + 3, 0, 0);   // next timestep's input row
    }
    __syncthreads();

    // Preload ALL A fragments for this step: one dscnt wait, then WMMAs back-to-back.
    v16h Afr[8];
#pragma unroll
    for (int kt=0; kt<8; kt++) Afr[kt] = load_a_frag(hcur, 272, kt);

    v8f acc[6];
#pragma unroll
    for (int i=0;i<6;i++) acc[i] = zero8();
#pragma unroll
    for (int kt=0; kt<8; kt++){
#pragma unroll
      for (int nt=0; nt<6; nt++){
        v16h Bf = load_b_frag(whh, (wave*6+nt)*8 + kt);
        acc[nt] = __builtin_amdgcn_wmma_f32_16x16x32_f16(
            false, Afr[kt], false, Bf, (short)0, acc[nt], false, false);
      }
    }
    {
      int l15 = lane & 15, mb = (lane>>4)*8;
#pragma unroll
      for (int nt=0; nt<6; nt++){
        int n = (wave*6+nt)*16 + l15;
#pragma unroll
        for (int r=0;r<8;r++) hp[(mb+r)*H3 + n] = (_Float16)acc[nt][r];
      }
    }
    __syncthreads();

#pragma unroll 4
    for (int m=0;m<16;m++){
      float x0=sx[m*4+0], x1=sx[m*4+1], x2=sx[m*4+2];
      float xr = x0*w_r0 + x1*w_r1 + x2*w_r2 + bi_r;
      float xz = x0*w_z0 + x1*w_z1 + x2*w_z2 + bi_z;
      float xn = x0*w_n0 + x1*w_n1 + x2*w_n2 + bi_n;
      float hr = (float)hp[m*H3 + j]       + bh_r;
      float hz = (float)hp[m*H3 + 256 + j] + bh_z;
      float hn = (float)hp[m*H3 + 512 + j] + bh_n;
      float rg = sigmoidf_(xr + hr);
      float zg = sigmoidf_(xz + hz);
      float ng = tanhf(xn + rg*hn);
      float hprev = (float)hcur[m*272 + j];
      float hnew = (1.0f - zg)*ng + zg*hprev;
      hcur[m*272 + j] = (_Float16)hnew;
      out[((size_t)(b0+m)*L_SZ + t)*H_SZ + j] = (_Float16)hnew;
    }
    __syncthreads();
  }
}

// ---- GRU layer 1: x-projection (K=256) + hidden GEMM, both WMMA, fused per step ----
__global__ void __launch_bounds__(256) gru_l1_kernel(
    const _Float16* __restrict__ xin_a, const _Float16* __restrict__ xin_b,
    const _Float16* __restrict__ wih_a, const _Float16* __restrict__ wih_b,
    const _Float16* __restrict__ whh_a, const _Float16* __restrict__ whh_b,
    const float* __restrict__ bih_a, const float* __restrict__ bhh_a,
    const float* __restrict__ bih_b, const float* __restrict__ bhh_b,
    _Float16* __restrict__ out_a, _Float16* __restrict__ out_b2)
{
  const int g = blockIdx.y;
  const _Float16* xin = g ? xin_b : xin_a;
  const _Float16* wih = g ? wih_b : wih_a;
  const _Float16* whh = g ? whh_b : whh_a;
  const float* bih    = g ? bih_b : bih_a;
  const float* bhh    = g ? bhh_b : bhh_a;
  _Float16* out       = g ? out_b2 : out_a;

  __shared__ _Float16 hp[16*H3];     // x@wih^T + h@whh^T for r,z cols; h-part only for n cols
  __shared__ _Float16 xnb[16*H_SZ];  // x-part of n-gate columns
  __shared__ _Float16 hcur[16*272];
  __shared__ _Float16 xbuf[16*272];
  __shared__ float sbih[H3];
  __shared__ float sbhh[H3];

  const int tid  = threadIdx.x;
  const int wave = tid >> 5;
  const int lane = tid & 31;
  const int b0   = blockIdx.x * 16;

  for (int i = tid; i < H3; i += 256){ sbih[i] = bih[i]; sbhh[i] = bhh[i]; }
  for (int i = tid; i < 16*272; i += 256){ hcur[i] = (_Float16)0.0f; xbuf[i] = (_Float16)0.0f; }
  __syncthreads();

  const int j = tid;
  const float bi_r = sbih[j], bi_z = sbih[256+j], bi_n = sbih[512+j];
  const float bh_r = sbhh[j], bh_z = sbhh[256+j], bh_n = sbhh[512+j];

  for (int t = 0; t < L_SZ; ++t){
    { // stage layer-0 output row tile (16 rows x 256 f16) into LDS
      int m = tid >> 4; int j0 = (tid & 15) << 4;
      const _Float16* srcp = xin + ((size_t)(b0+m)*L_SZ + t)*H_SZ + j0;
      *(v16h*)(xbuf + m*272 + j0) = *(const v16h*)srcp;
      __builtin_prefetch(srcp + H_SZ, 0, 0);   // next timestep's row
    }
    __syncthreads();

    // Preload all A fragments (h and x) up front: single wait, then 96 WMMAs.
    v16h Ah[8], Ax[8];
#pragma unroll
    for (int kt=0; kt<8; kt++){
      Ah[kt] = load_a_frag(hcur, 272, kt);
      Ax[kt] = load_a_frag(xbuf, 272, kt);
    }

    v8f ah[6], ax[6];
#pragma unroll
    for (int i=0;i<6;i++){ ah[i] = zero8(); ax[i] = zero8(); }
#pragma unroll
    for (int kt=0; kt<8; kt++){
#pragma unroll
      for (int nt=0; nt<6; nt++){
        int frag = (wave*6+nt)*8 + kt;
        v16h Bh = load_b_frag(whh, frag);
        ah[nt] = __builtin_amdgcn_wmma_f32_16x16x32_f16(
            false, Ah[kt], false, Bh, (short)0, ah[nt], false, false);
        v16h Bx = load_b_frag(wih, frag);
        ax[nt] = __builtin_amdgcn_wmma_f32_16x16x32_f16(
            false, Ax[kt], false, Bx, (short)0, ax[nt], false, false);
      }
    }
    {
      int l15 = lane & 15, mb = (lane>>4)*8;
#pragma unroll
      for (int nt=0; nt<6; nt++){
        int ntile = wave*6+nt;
        int n = ntile*16 + l15;
        if (ntile < 32){ // r,z columns: combine x+h (biases added in gate phase)
#pragma unroll
          for (int r=0;r<8;r++) hp[(mb+r)*H3 + n] = (_Float16)(ah[nt][r] + ax[nt][r]);
        } else {         // n columns: keep x and h parts separate
#pragma unroll
          for (int r=0;r<8;r++){
            hp[(mb+r)*H3 + n]          = (_Float16)ah[nt][r];
            xnb[(mb+r)*H_SZ + (n-512)] = (_Float16)ax[nt][r];
          }
        }
      }
    }
    __syncthreads();

#pragma unroll 4
    for (int m=0;m<16;m++){
      float sr = (float)hp[m*H3 + j]       + bi_r + bh_r;
      float sz = (float)hp[m*H3 + 256 + j] + bi_z + bh_z;
      float xn = (float)xnb[m*H_SZ + j]    + bi_n;
      float hn = (float)hp[m*H3 + 512 + j] + bh_n;
      float rg = sigmoidf_(sr);
      float zg = sigmoidf_(sz);
      float ng = tanhf(xn + rg*hn);
      float hprev = (float)hcur[m*272 + j];
      float hnew = (1.0f - zg)*ng + zg*hprev;
      hcur[m*272 + j] = (_Float16)hnew;
      out[((size_t)(b0+m)*L_SZ + t)*H_SZ + j] = (_Float16)hnew;
    }
    __syncthreads();
  }
}

// ---- sliding-window attention + FC fusion (rows t>=5), WMMA GEMM for the FC ----
__global__ void __launch_bounds__(256) fuse_kernel(
    const _Float16* __restrict__ l1o_a, const _Float16* __restrict__ l1o_b,
    const _Float16* __restrict__ fcw, const float* __restrict__ fc_b,
    const float* __restrict__ attn_w,
    _Float16* __restrict__ fu_a, _Float16* __restrict__ fu_b)
{
  const _Float16* src = blockIdx.y ? l1o_b : l1o_a;
  _Float16*       dst = blockIdx.y ? fu_b  : fu_a;

  __shared__ _Float16 At[16*528];      // [ctx | q] tile, 32B-aligned padded stride
  __shared__ float part[16][16][6];
  __shared__ float aw[16][NWIN];
  __shared__ float swq[2*H_SZ];
  __shared__ float sfb[H_SZ];

  const int tid  = threadIdx.x;
  const int wave = tid >> 5;
  const int lane = tid & 31;

  for (int i = tid; i < 2*H_SZ; i += 256) swq[i] = attn_w[i];
  for (int i = tid; i < H_SZ;   i += 256) sfb[i] = fc_b[i];
  __syncthreads();

  const int NT = (B_SZ*(L_SZ-NWIN))/16;  // 4056 row tiles
  for (int tile = blockIdx.x; tile < NT; tile += gridDim.x){
    const int R0 = tile*16;

    { // stage q into right half of A (one v16h per thread)
      int m = tid >> 4; int j0 = (tid & 15) << 4;
      int R = R0+m; int bb = R/(L_SZ-NWIN); int tt = R - bb*(L_SZ-NWIN) + NWIN;
      const _Float16* srcp = src + ((size_t)bb*L_SZ + tt)*H_SZ + j0;
      *(v16h*)(At + m*528 + 256 + j0) = *(const v16h*)srcp;
    }
    __syncthreads();

    { // partial attention dots (16 threads per row, v8h-vectorized)
      int m = tid >> 4, c = tid & 15;
      int R = R0+m; int bb = R/(L_SZ-NWIN); int tt = R - bb*(L_SZ-NWIN) + NWIN;
      const _Float16* wrow = src + ((size_t)bb*L_SZ + (tt-NWIN))*H_SZ;
      float pq=0.f, p0=0.f, p1=0.f, p2=0.f, p3=0.f, p4=0.f;
#pragma unroll
      for (int h8=0; h8<2; h8++){
        int jj0 = c*16 + h8*8;
        v8h qv = *(const v8h*)(At + m*528 + 256 + jj0);
        v8h w0 = *(const v8h*)(wrow + jj0);
        v8h w1 = *(const v8h*)(wrow + 256  + jj0);
        v8h w2 = *(const v8h*)(wrow + 512  + jj0);
        v8h w3 = *(const v8h*)(wrow + 768  + jj0);
        v8h w4 = *(const v8h*)(wrow + 1024 + jj0);
#pragma unroll
        for (int u=0; u<8; u++){
          float wqv = swq[jj0+u], wkv = swq[256+jj0+u];
          pq += (float)qv[u]*wqv;
          p0 += (float)w0[u]*wkv;
          p1 += (float)w1[u]*wkv;
          p2 += (float)w2[u]*wkv;
          p3 += (float)w3[u]*wkv;
          p4 += (float)w4[u]*wkv;
        }
      }
      part[m][c][0]=pq; part[m][c][1]=p0; part[m][c][2]=p1;
      part[m][c][3]=p2; part[m][c][4]=p3; part[m][c][5]=p4;
    }
    __syncthreads();

    if (tid < 16){ // softmax over W=5
      int m = tid;
      float q=0.f, s0=0.f,s1=0.f,s2=0.f,s3=0.f,s4=0.f;
      for (int c=0;c<16;c++){
        q+=part[m][c][0]; s0+=part[m][c][1]; s1+=part[m][c][2];
        s2+=part[m][c][3]; s3+=part[m][c][4]; s4+=part[m][c][5];
      }
      float sc[NWIN] = {s0+q, s1+q, s2+q, s3+q, s4+q};
      float mx = sc[0];
      for (int k=1;k<NWIN;k++) mx = fmaxf(mx, sc[k]);
      float e[NWIN], sum=0.f;
      for (int k=0;k<NWIN;k++){ e[k]=__expf(sc[k]-mx); sum+=e[k]; }
      float inv = 1.0f/sum;
      for (int k=0;k<NWIN;k++) aw[m][k] = e[k]*inv;
    }
    __syncthreads();

    // ctx into left half of A (v8h-vectorized, 2 chunks per thread)
    for (int idx=tid; idx<512; idx+=256){
      int m = idx>>5, jj0 = (idx & 31)*8;
      int R = R0+m; int bb = R/(L_SZ-NWIN); int tt = R - bb*(L_SZ-NWIN) + NWIN;
      const _Float16* wr = src + ((size_t)bb*L_SZ + (tt-NWIN))*H_SZ + jj0;
      v8h a0v = *(const v8h*)wr;
      v8h a1v = *(const v8h*)(wr + 256);
      v8h a2v = *(const v8h*)(wr + 512);
      v8h a3v = *(const v8h*)(wr + 768);
      v8h a4v = *(const v8h*)(wr + 1024);
      float c0 = aw[m][0], c1 = aw[m][1], c2 = aw[m][2], c3 = aw[m][3], c4 = aw[m][4];
      v8h ov;
#pragma unroll
      for (int u=0; u<8; u++){
        ov[u] = (_Float16)(c0*(float)a0v[u] + c1*(float)a1v[u] + c2*(float)a2v[u]
                         + c3*(float)a3v[u] + c4*(float)a4v[u]);
      }
      *(v8h*)(At + m*528 + jj0) = ov;
    }
    __syncthreads();

    // FC GEMM: [16 x 512] x [512 x 256] -- preload A frags, then WMMA stream
    v16h Afr[16];
#pragma unroll
    for (int kt=0; kt<16; kt++) Afr[kt] = load_a_frag(At, 528, kt);
    v8f a0 = zero8(), a1 = zero8();
#pragma unroll
    for (int kt=0; kt<16; kt++){
      v16h B0 = load_b_frag(fcw, (wave*2+0)*16 + kt);
      a0 = __builtin_amdgcn_wmma_f32_16x16x32_f16(false, Afr[kt], false, B0, (short)0, a0, false, false);
      v16h B1 = load_b_frag(fcw, (wave*2+1)*16 + kt);
      a1 = __builtin_amdgcn_wmma_f32_16x16x32_f16(false, Afr[kt], false, B1, (short)0, a1, false, false);
    }
    {
      int l15 = lane & 15, mb = (lane>>4)*8;
      int n0 = (wave*2+0)*16 + l15;
      int n1 = (wave*2+1)*16 + l15;
#pragma unroll
      for (int r=0;r<8;r++){
        int R = R0 + mb + r; int bb = R/(L_SZ-NWIN); int tt = R - bb*(L_SZ-NWIN) + NWIN;
        size_t rowoff = ((size_t)bb*L_SZ + tt)*H_SZ;
        dst[rowoff + n0] = (_Float16)(a0[r] + sfb[n0]);
        dst[rowoff + n1] = (_Float16)(a1[r] + sfb[n1]);
      }
    }
    __syncthreads();
  }
}

// ---- final head: sigmoid( out_w[0:256].rnn1[b,t] + out_w[256:512].rnn2[b,min(t+10,511)] + b ) ----
__global__ void __launch_bounds__(256) final_kernel(
    const _Float16* __restrict__ l1o1, const _Float16* __restrict__ fu1,
    const _Float16* __restrict__ fu2,
    const float* __restrict__ out_w, const float* __restrict__ out_b,
    float* __restrict__ out)
{
  const int wave = threadIdx.x >> 5;
  const int lane = threadIdx.x & 31;
  const int idx = blockIdx.x*8 + wave;           // b*512 + t
  const int b = idx >> 9, t = idx & 511;
  const _Float16* r1 = (t < NWIN ? l1o1 : fu1) + ((size_t)b*L_SZ + t)*H_SZ;
  const int td = (t + 10 > L_SZ-1) ? (L_SZ-1) : (t + 10);   // always >= NWIN -> fused region
  const _Float16* r2 = fu2 + ((size_t)b*L_SZ + td)*H_SZ;

  const int j0 = lane*8;   // contiguous 8-element chunk per lane
  v8h r1v = *(const v8h*)(r1 + j0);
  v8h r2v = *(const v8h*)(r2 + j0);
  float4 wa0 = *(const float4*)(out_w + j0);
  float4 wa1 = *(const float4*)(out_w + j0 + 4);
  float4 wb0 = *(const float4*)(out_w + 256 + j0);
  float4 wb1 = *(const float4*)(out_w + 256 + j0 + 4);
  float s = wa0.x*(float)r1v[0] + wa0.y*(float)r1v[1] + wa0.z*(float)r1v[2] + wa0.w*(float)r1v[3]
          + wa1.x*(float)r1v[4] + wa1.y*(float)r1v[5] + wa1.z*(float)r1v[6] + wa1.w*(float)r1v[7]
          + wb0.x*(float)r2v[0] + wb0.y*(float)r2v[1] + wb0.z*(float)r2v[2] + wb0.w*(float)r2v[3]
          + wb1.x*(float)r2v[4] + wb1.y*(float)r2v[5] + wb1.z*(float)r2v[6] + wb1.w*(float)r2v[7];
#pragma unroll
  for (int off=16; off; off>>=1) s += __shfl_xor(s, off, 32);
  if (lane == 0) out[idx] = sigmoidf_(s + out_b[0]);
}

extern "C" void kernel_launch(void* const* d_in, const int* in_sizes, int n_in,
                              void* d_out, int out_size, void* d_ws, size_t ws_size,
                              hipStream_t stream)
{
  const float* x       = (const float*)d_in[0];
  const float* g1_wih0 = (const float*)d_in[1];
  const float* g1_whh0 = (const float*)d_in[2];
  const float* g1_bih0 = (const float*)d_in[3];
  const float* g1_bhh0 = (const float*)d_in[4];
  const float* g1_wih1 = (const float*)d_in[5];
  const float* g1_whh1 = (const float*)d_in[6];
  const float* g1_bih1 = (const float*)d_in[7];
  const float* g1_bhh1 = (const float*)d_in[8];
  const float* g2_wih0 = (const float*)d_in[9];
  const float* g2_whh0 = (const float*)d_in[10];
  const float* g2_bih0 = (const float*)d_in[11];
  const float* g2_bhh0 = (const float*)d_in[12];
  const float* g2_wih1 = (const float*)d_in[13];
  const float* g2_whh1 = (const float*)d_in[14];
  const float* g2_bih1 = (const float*)d_in[15];
  const float* g2_bhh1 = (const float*)d_in[16];
  const float* fc_w    = (const float*)d_in[17];
  const float* fc_b    = (const float*)d_in[18];
  const float* attn_w  = (const float*)d_in[19];
  const float* out_w   = (const float*)d_in[20];
  const float* out_b   = (const float*)d_in[21];
  (void)in_sizes; (void)n_in; (void)out_size; (void)ws_size;

  char* p = (char*)d_ws;
  auto alloc = [&](size_t bytes)->void*{
    void* r = (void*)p; p += (bytes + 255) & ~(size_t)255; return r;
  };
  const size_t WT  = (size_t)H3*H_SZ*sizeof(_Float16);            // 384 KB
  const size_t FCT = (size_t)H_SZ*(2*H_SZ)*sizeof(_Float16);      // 256 KB
  const size_t SEQ = (size_t)B_SZ*L_SZ*H_SZ*sizeof(_Float16);     // 32 MB

  _Float16* whh0_t1 = (_Float16*)alloc(WT);
  _Float16* whh0_t2 = (_Float16*)alloc(WT);
  _Float16* wih1_t1 = (_Float16*)alloc(WT);
  _Float16* wih1_t2 = (_Float16*)alloc(WT);
  _Float16* whh1_t1 = (_Float16*)alloc(WT);
  _Float16* whh1_t2 = (_Float16*)alloc(WT);
  _Float16* fcw_t   = (_Float16*)alloc(FCT);
  _Float16* l0o1 = (_Float16*)alloc(SEQ);
  _Float16* l0o2 = (_Float16*)alloc(SEQ);
  _Float16* l1o1 = (_Float16*)alloc(SEQ);
  _Float16* l1o2 = (_Float16*)alloc(SEQ);
  _Float16* fu1  = (_Float16*)alloc(SEQ);
  _Float16* fu2  = (_Float16*)alloc(SEQ);

  // weight conversions (tiny; all L2-resident afterwards)
  convert_tiled_kernel<<<768, 256, 0, stream>>>(g1_whh0, whh0_t1, H3, H_SZ);
  convert_tiled_kernel<<<768, 256, 0, stream>>>(g2_whh0, whh0_t2, H3, H_SZ);
  convert_tiled_kernel<<<768, 256, 0, stream>>>(g1_wih1, wih1_t1, H3, H_SZ);
  convert_tiled_kernel<<<768, 256, 0, stream>>>(g2_wih1, wih1_t2, H3, H_SZ);
  convert_tiled_kernel<<<768, 256, 0, stream>>>(g1_whh1, whh1_t1, H3, H_SZ);
  convert_tiled_kernel<<<768, 256, 0, stream>>>(g2_whh1, whh1_t2, H3, H_SZ);
  convert_tiled_kernel<<<512, 256, 0, stream>>>(fc_w, fcw_t, H_SZ, 2*H_SZ);

  // GRU layer 0 (both GRUs, all batch tiles concurrent)
  gru_l0_kernel<<<dim3(B_SZ/16, 2), 256, 0, stream>>>(
      x, g1_wih0, g2_wih0, whh0_t1, whh0_t2,
      g1_bih0, g1_bhh0, g2_bih0, g2_bhh0, l0o1, l0o2);

  // GRU layer 1 (x-projection fused into the recurrent step)
  gru_l1_kernel<<<dim3(B_SZ/16, 2), 256, 0, stream>>>(
      l0o1, l0o2, wih1_t1, wih1_t2, whh1_t1, whh1_t2,
      g1_bih1, g1_bhh1, g2_bih1, g2_bhh1, l1o1, l1o2);

  // sliding-window attention + FC fusion
  fuse_kernel<<<dim3(256, 2), 256, 0, stream>>>(
      l1o1, l1o2, fcw_t, fc_b, attn_w, fu1, fu2);

  // delayed-combine sigmoid head
  final_kernel<<<(B_SZ*L_SZ)/8, 256, 0, stream>>>(
      l1o1, fu1, fu2, out_w, out_b, (float*)d_out);
}
